// CriticValue_32160715112722
// MI455X (gfx1250) — matrix operand order repaired and checked by reading, not compiled, not run on hardware
//
#include <hip/hip_runtime.h>

// Problem constants (from setup_inputs): B=4096, N=64, OD=128, A=16, H=64, M=32, C=80
#define B_TOT 4096
#define N_AG  64
#define OD_   128
#define A_DIM 16

// Packed-weight plane offsets in d_ws (dwords)
#define W1H 0
#define W1L 4096
#define W2H 8192
#define W2L 10240
#define CW1H 12288
#define CW1L 13824
#define SW1H 15360
#define SW1L 16896
#define CW2H 18432
#define CW2L 18688
#define SW2H 18944
#define SW2L 19200
#define PFH 19456
#define PFL 19712
#define WS_DWORDS 19968

typedef __attribute__((ext_vector_type(16))) __bf16    v16bf;
typedef __attribute__((ext_vector_type(8)))  float     v8f;
typedef __attribute__((ext_vector_type(4)))  unsigned  v4u;

static __device__ __forceinline__ unsigned short f2bfbits(float x) { // RNE
  union { float f; unsigned u; } a; a.f = x;
  return (unsigned short)((a.u + 0x7fffu + ((a.u >> 16) & 1u)) >> 16);
}
static __device__ __forceinline__ float bfbits2f(unsigned short s) {
  union { unsigned u; float f; } a; a.u = ((unsigned)s) << 16; return a.f;
}
// Split two f32 into bf16 hi/lo pairs packed as dwords (elem0 in [15:0]).
static __device__ __forceinline__ void split_pack(float x0, float x1,
                                                  unsigned& hdw, unsigned& ldw) {
  const unsigned short h0 = f2bfbits(x0), h1 = f2bfbits(x1);
  const float l0 = x0 - bfbits2f(h0), l1 = x1 - bfbits2f(h1);
  hdw = (unsigned)h0 | ((unsigned)h1 << 16);
  ldw = (unsigned)f2bfbits(l0) | ((unsigned)f2bfbits(l1) << 16);
}

struct Frag { v16bf h, l; };
union VU { v4u q[2]; unsigned u[8]; v16bf v; };

// Fragment-major B panel addressing: S(k2,n) = (k2>>3)*512 + n*8 + (k2&7).

// A fragment (16x32 bf16) from packed planes: plane[m*lda2 + k2] holds pair (2k2, 2k2+1).
// dword v -> k2 = k20 + (v>>2)*8 + half*4 + (v&3): two contiguous quads -> 2x b128.
static __device__ __forceinline__ Frag load_A_pl(const unsigned* __restrict__ aH,
                                                 const unsigned* __restrict__ aL,
                                                 int lda2, int row0, int k20, int lane) {
  const int m = row0 + (lane & 15);
  const int half = lane >> 4;
  const int base = m * lda2 + k20 + (half << 2);
  VU H, L;
  H.q[0] = *(const v4u*)(aH + base);
  H.q[1] = *(const v4u*)(aH + base + 8);
  L.q[0] = *(const v4u*)(aL + base);
  L.q[1] = *(const v4u*)(aL + base + 8);
  Frag f; f.h = H.v; f.l = L.v; return f;
}

// B fragment (32x16 bf16) from fragment-major LDS planes: dword v -> k2 = k20 + (v>>2)*8 + half*4 + (v&3) -> 2x b128.
static __device__ __forceinline__ Frag load_B_fm(const unsigned* __restrict__ bH,
                                                 const unsigned* __restrict__ bL,
                                                 int k20, int col0, int lane) {
  const int n = col0 + (lane & 15);
  const int khalf = lane >> 4;
  const int base = (k20 >> 3) * 512 + n * 8 + (khalf << 2);
  VU H, L;
  H.q[0] = *(const v4u*)(bH + base);
  H.q[1] = *(const v4u*)(bH + base + 512);
  L.q[0] = *(const v4u*)(bL + base);
  L.q[1] = *(const v4u*)(bL + base + 512);
  Frag f; f.h = H.v; f.l = L.v; return f;
}

// A fragment from f32 with on-the-fly split (only for cv as A operand of mix).
static __device__ __forceinline__ Frag load_A_f32(const float* __restrict__ base, int ld,
                                                  int row0, int k0, int lane) {
  const int m = row0 + (lane & 15);
  const int half = lane >> 4;
  const float* rp = base + m * ld;
  VU H, L;
#pragma unroll
  for (int v = 0; v < 8; ++v) {
    const int k = k0 + ((v >> 2) << 4) + (half << 3) + ((v & 3) << 1);
    split_pack(rp[k], rp[k + 1], H.u[v], L.u[v]);
  }
  Frag f; f.h = H.v; f.l = L.v; return f;
}

// 3-term bf16-split product: near-fp32 via v_wmma_f32_16x16x32_bf16 x3.
static __device__ __forceinline__ v8f wmma3(const Frag& a, const Frag& b, v8f c) {
  c = __builtin_amdgcn_wmma_f32_16x16x32_bf16(false, a.h, false, b.h, (short)0, c, false, false);
  c = __builtin_amdgcn_wmma_f32_16x16x32_bf16(false, a.h, false, b.l, (short)0, c, false, false);
  c = __builtin_amdgcn_wmma_f32_16x16x32_bf16(false, a.l, false, b.h, (short)0, c, false, false);
  return c;
}

template <int NCT>
static __device__ __forceinline__ void init_acc(v8f (&acc)[NCT], const float* bias,
                                                int row0, int lane) {
  const int half = lane >> 4;
#pragma unroll
  for (int ct = 0; ct < NCT; ++ct)
#pragma unroll
    for (int r = 0; r < 8; ++r)
      acc[ct][r] = bias ? bias[row0 + 8 * half + r] : 0.0f;
}

// Planes-A (global packed) x FM-planes-B (LDS) GEMM accumulate.
template <int NCT, int NK>
static __device__ __forceinline__ void gemm_pp(v8f (&acc)[NCT],
    const unsigned* aH, const unsigned* aL, int lda2, int row0,
    const unsigned* bH, const unsigned* bL, int col0, int lane) {
#pragma unroll
  for (int kc = 0; kc < NK; ++kc) {
    const Frag a = load_A_pl(aH, aL, lda2, row0, kc * 16, lane);
#pragma unroll
    for (int ct = 0; ct < NCT; ++ct) {
      const Frag b = load_B_fm(bH, bL, kc * 16, col0 + ct * 16, lane);
      acc[ct] = wmma3(a, b, acc[ct]);
    }
  }
}

// f32-A x FM-planes-B GEMM accumulate.
template <int NCT, int NK>
static __device__ __forceinline__ void gemm_fp(v8f (&acc)[NCT],
    const float* aF, int lda, int row0,
    const unsigned* bH, const unsigned* bL, int col0, int lane) {
#pragma unroll
  for (int kc = 0; kc < NK; ++kc) {
    const Frag a = load_A_f32(aF, lda, row0, kc * 32, lane);
#pragma unroll
    for (int ct = 0; ct < NCT; ++ct) {
      const Frag b = load_B_fm(bH, bL, kc * 16, col0 + ct * 16, lane);
      acc[ct] = wmma3(a, b, acc[ct]);
    }
  }
}

// Store D as fragment-major packed bf16 hi/lo B-planes: one b128 per plane per tile.
template <int NCT>
static __device__ __forceinline__ void store_packed(const v8f (&acc)[NCT],
    unsigned* pH, unsigned* pL, int k2base, int row0, int col0, bool relu, int lane) {
  const int half = lane >> 4;
  const int n0 = lane & 15;
  const int blk = (k2base + (row0 >> 1)) >> 3;  // k2base, row0/2 are multiples of 8
#pragma unroll
  for (int ct = 0; ct < NCT; ++ct) {
    const int n = col0 + ct * 16 + n0;
    v4u hq, lq;
#pragma unroll
    for (int i = 0; i < 4; ++i) {
      float x0 = acc[ct][2 * i], x1 = acc[ct][2 * i + 1];
      if (relu) { x0 = fmaxf(x0, 0.0f); x1 = fmaxf(x1, 0.0f); }
      unsigned hdw, ldw;
      split_pack(x0, x1, hdw, ldw);
      hq[i] = hdw; lq[i] = ldw;
    }
    const int idx = blk * 512 + n * 8 + 4 * half;
    *(v4u*)(pH + idx) = hq;
    *(v4u*)(pL + idx) = lq;
  }
}

template <int NCT>
static __device__ __forceinline__ void store_f32(const v8f (&acc)[NCT],
    float* D, int ldd, int row0, int col0, bool relu, int lane) {
  const int half = lane >> 4;
  const int n0 = lane & 15;
#pragma unroll
  for (int ct = 0; ct < NCT; ++ct) {
    const int n = col0 + ct * 16 + n0;
#pragma unroll
    for (int r = 0; r < 8; ++r) {
      float v = acc[ct][r];
      if (relu) v = fmaxf(v, 0.0f);
      D[(row0 + 8 * half + r) * ldd + n] = v;
    }
  }
}

// ---------------- weight pre-pack kernel (runs once per launch) ----------------
static __device__ __forceinline__ void pack_w_g(const float* __restrict__ w, int K,
                                                int M, int K2pad,
                                                unsigned* pH, unsigned* pL,
                                                int tid, int stride) {
  for (int i = tid; i < M * K2pad; i += stride) {
    const int m = i / K2pad, k2 = i - m * K2pad;
    const int k = 2 * k2;
    const float x0 = (k < K) ? w[m * K + k] : 0.0f;
    const float x1 = (k + 1 < K) ? w[m * K + k + 1] : 0.0f;
    split_pack(x0, x1, pH[i], pL[i]);
  }
}

__global__ __launch_bounds__(256) void pack_weights(
    const float* __restrict__ enc_w1, const float* __restrict__ enc_w2,
    const float* __restrict__ cv_w1, const float* __restrict__ sv_w1,
    const float* __restrict__ cv_w2, const float* __restrict__ sv_w2,
    const float* __restrict__ pf_w, unsigned* __restrict__ ws) {
  const int tid = blockIdx.x * 256 + threadIdx.x;
  const int stride = gridDim.x * 256;
  pack_w_g(enc_w1, 128, 64, 64, ws + W1H, ws + W1L, tid, stride);
  pack_w_g(enc_w2, 64, 64, 32, ws + W2H, ws + W2L, tid, stride);
  pack_w_g(cv_w1, 80, 32, 48, ws + CW1H, ws + CW1L, tid, stride);
  pack_w_g(sv_w1, 80, 32, 48, ws + SW1H, ws + SW1L, tid, stride);
  pack_w_g(cv_w2, 32, 16, 16, ws + CW2H, ws + CW2L, tid, stride);
  pack_w_g(sv_w2, 32, 16, 16, ws + SW2H, ws + SW2L, tid, stride);
  pack_w_g(pf_w, 32, 16, 16, ws + PFH, ws + PFL, tid, stride);
}

// ---------------- main fused kernel: one block per batch element ----------------
__global__ __launch_bounds__(128) void critic_fused(
    const float* __restrict__ obs, const float* __restrict__ actions,
    const float* __restrict__ relations, const unsigned* __restrict__ wsp,
    const float* __restrict__ enc_b1, const float* __restrict__ enc_b2,
    const float* __restrict__ cv_b1, const float* __restrict__ cv_b2,
    const float* __restrict__ sv_b1, const float* __restrict__ sv_b2,
    const float* __restrict__ pf_b, float* __restrict__ out) {
  // sm1: obs FM planes (4096+4096) -> sa FM planes (3072+3072)
  //      -> { relH 0..2047, relL 2048..4095, cv f32 4096..5119,
  //           catH 5120..6143, catL 6144..7167, post f32 7168..8191 }
  __shared__ unsigned sm1[8192];
  // sm3: h FM planes (2048+2048) -> { cv1 H/L 0/1024, sv1 H/L 2048/3072 }
  __shared__ unsigned sm3[4096];

  const int b = blockIdx.x;
  const int tid = threadIdx.x;
  const int w = tid >> 5;
  const int lane = tid & 31;

  // ---- Phase 1: pack obs[b] into FM B-planes (coalesced reads, n fastest) ----
  {
    const float* ob = obs + (size_t)b * (OD_ * N_AG);
    for (int i = tid; i < 4096; i += 128) {
      const int k2 = i >> 6, n = i & 63;
      const int S = ((k2 >> 3) << 9) + (n << 3) + (k2 & 7);
      split_pack(ob[k2 * 128 + n], ob[k2 * 128 + 64 + n], sm1[S], sm1[4096 + S]);
    }
  }
  __syncthreads();

  // ---- Phase 2: h = relu(enc_w1 @ obs + b1) -> h FM planes ----
  {
    v8f acc[4];
    init_acc(acc, enc_b1, 16 * w, lane);
    gemm_pp<4, 4>(acc, wsp + W1H, wsp + W1L, 64, 16 * w, sm1, sm1 + 4096, 0, lane);
    store_packed(acc, sm3, sm3 + 2048, 0, 16 * w, 0, true, lane);
  }
  __syncthreads();

  // ---- Phase 3: sa = [enc_w2 @ h + b2 ; actions ; zeros] -> sa FM planes ----
  {
    v8f acc[4];
    init_acc(acc, enc_b2, 16 * w, lane);
    gemm_pp<4, 2>(acc, wsp + W2H, wsp + W2L, 32, 16 * w, sm3, sm3 + 2048, 0, lane);
    store_packed(acc, sm1, sm1 + 3072, 0, 16 * w, 0, false, lane);
  }
  for (int i = tid; i < 512; i += 128) {
    const int v = i & 7, n = (i >> 3) & 63;   // k2 = 32 + v -> action rows 2v, 2v+1
    const float* ap = actions + ((size_t)n * B_TOT + b) * A_DIM + 2 * v;
    split_pack(ap[0], ap[1], sm1[2048 + i], sm1[5120 + i]);
  }
  for (int i = tid; i < 512; i += 128) { sm1[2560 + i] = 0u; sm1[5632 + i] = 0u; }
  __syncthreads();

  // ---- Phase 4: cv1/sv1 = relu(W1 @ sa + b1) -> FM planes in sm3 ----
  {
    const unsigned* aH = wsp + ((w < 2) ? CW1H : SW1H);
    const unsigned* aL = wsp + ((w < 2) ? CW1L : SW1L);
    const float* bb = (w < 2) ? cv_b1 : sv_b1;
    unsigned* oH = (w < 2) ? sm3 : (sm3 + 2048);
    unsigned* oL = (w < 2) ? (sm3 + 1024) : (sm3 + 3072);
    v8f acc[4];
    init_acc(acc, bb, 16 * (w & 1), lane);
    gemm_pp<4, 3>(acc, aH, aL, 48, 16 * (w & 1), sm1, sm1 + 3072, 0, lane);
    store_packed(acc, oH, oL, 0, 16 * (w & 1), 0, true, lane);
  }
  __syncthreads();

  // ---- Phase 5: rel FM planes; cv = cvW2@cv1+b (f32); sv -> cat rows 16-31 ----
  for (int i = tid; i < 2048; i += 128) {
    const int k2 = ((i >> 9) << 3) | (i & 7);
    const int n = (i >> 3) & 63;
    const size_t rb = ((size_t)n * B_TOT + b) * (N_AG - 1);
    float x[2];
#pragma unroll
    for (int e = 0; e < 2; ++e) {
      const int j = 2 * k2 + e;
      x[e] = (j == n) ? 1.0f : relations[rb + ((j < n) ? j : (j - 1))];
    }
    split_pack(x[0], x[1], sm1[i], sm1[2048 + i]);
  }
  if (w < 2) {
    v8f acc[2];
    init_acc(acc, cv_b2, 0, lane);
    gemm_pp<2, 1>(acc, wsp + CW2H, wsp + CW2L, 16, 0, sm3, sm3 + 1024,
                  32 * (w & 1), lane);
    store_f32(acc, (float*)(sm1 + 4096), 64, 0, 32 * (w & 1), false, lane);
  } else {
    v8f acc[2];
    init_acc(acc, sv_b2, 0, lane);
    gemm_pp<2, 1>(acc, wsp + SW2H, wsp + SW2L, 16, 0, sm3 + 2048, sm3 + 3072,
                  32 * (w & 1), lane);
    store_packed(acc, sm1 + 5120, sm1 + 6144, 8, 0, 32 * (w & 1), false, lane);
  }
  __syncthreads();

  // ---- Phase 6: cat rows 0-15 = cv @ rel (K=64) ----
  {
    v8f acc[1];
    init_acc(acc, (const float*)nullptr, 0, lane);
    gemm_fp<1, 2>(acc, (const float*)(sm1 + 4096), 64, 0, sm1, sm1 + 2048, 16 * w, lane);
    store_packed(acc, sm1 + 5120, sm1 + 6144, 0, 0, 16 * w, false, lane);
  }
  __syncthreads();

  // ---- Phase 7: post = pf_w @ cat + pf_b (f32 for gather) ----
  {
    v8f acc[1];
    init_acc(acc, pf_b, 0, lane);
    gemm_pp<1, 1>(acc, wsp + PFH, wsp + PFL, 16, 0, sm1 + 5120, sm1 + 6144,
                  16 * w, lane);
    store_f32(acc, (float*)(sm1 + 7168), 64, 0, 16 * w, false, lane);
  }
  __syncthreads();

  // ---- Phase 8: q[n,b] = post[argmax_a actions[n,b,:]][n] ----
  if (tid < N_AG) {
    const int n = tid;
    const float* ap = actions + ((size_t)n * B_TOT + b) * A_DIM;
    const float* post = (const float*)(sm1 + 7168);
    float best = ap[0];
    int bi = 0;
#pragma unroll
    for (int a = 1; a < A_DIM; ++a) {
      const float v = ap[a];
      if (v > best) { best = v; bi = a; }
    }
    out[(size_t)n * B_TOT + b] = post[bi * 64 + n];
  }
}

extern "C" void kernel_launch(void* const* d_in, const int* in_sizes, int n_in,
                              void* d_out, int out_size, void* d_ws, size_t ws_size,
                              hipStream_t stream) {
  (void)in_sizes; (void)n_in; (void)out_size; (void)ws_size;
  const float* obs       = (const float*)d_in[0];
  const float* actions   = (const float*)d_in[1];
  const float* relations = (const float*)d_in[2];
  const float* enc_w1    = (const float*)d_in[3];
  const float* enc_b1    = (const float*)d_in[4];
  const float* enc_w2    = (const float*)d_in[5];
  const float* enc_b2    = (const float*)d_in[6];
  const float* cv_w1     = (const float*)d_in[7];
  const float* cv_b1     = (const float*)d_in[8];
  const float* cv_w2     = (const float*)d_in[9];
  const float* cv_b2     = (const float*)d_in[10];
  const float* sv_w1     = (const float*)d_in[11];
  const float* sv_b1     = (const float*)d_in[12];
  const float* sv_w2     = (const float*)d_in[13];
  const float* sv_b2     = (const float*)d_in[14];
  const float* pf_w      = (const float*)d_in[15];
  const float* pf_b      = (const float*)d_in[16];
  unsigned* ws = (unsigned*)d_ws;
  float* out = (float*)d_out;

  pack_weights<<<dim3(8), dim3(256), 0, stream>>>(
      enc_w1, enc_w2, cv_w1, sv_w1, cv_w2, sv_w2, pf_w, ws);

  critic_fused<<<dim3(B_TOT), dim3(128), 0, stream>>>(
      obs, actions, relations, ws, enc_b1, enc_b2, cv_b1, cv_b2,
      sv_b1, sv_b2, pf_b, out);
}
